// LlamaDifferentialAttentionBase_33200097198865
// MI455X (gfx1250) — matrix operand order repaired and checked by reading, not compiled, not run on hardware
//
#include <hip/hip_runtime.h>

// ---------------- problem constants ----------------
constexpr int S    = 2048;
constexpr int HID  = 2048;
constexpr int H    = 32;
constexpr int D    = 64;
constexpr int QKN  = 2 * HID;               // 4096
constexpr float SCALE      = 0.125f;        // 1/sqrt(64)
constexpr float EPS        = 1e-6f;
constexpr float LAM_INIT   = 0.7836057665f; // 0.8 - 0.6*exp(-3.6)
constexpr float OUT_SCALE  = 1.0f - LAM_INIT;
constexpr float LOG_THETA  = 9.2103403720f; // ln(10000)

// ---------------- types for WMMA ----------------
typedef __bf16 bf16_t;
typedef __attribute__((ext_vector_type(16))) __bf16 v16bf;
typedef __attribute__((ext_vector_type(8)))  __bf16 v8bf;
typedef __attribute__((ext_vector_type(8)))  float  v8f;

union V16 { v16bf v; v8bf h[2]; };
union VF8 { v8f  v; float f[8]; };

__device__ inline v8f wmma_bf16(const V16& a, const V16& b, const VF8& c) {
    return __builtin_amdgcn_wmma_f32_16x16x32_bf16(
        false, a.v, false, b.v, (short)0, c.v, false, false);
}

__device__ inline float redmax16(float v) {   // reduce across the 16-lane half
    v = fmaxf(v, __shfl_xor(v, 1, 32));
    v = fmaxf(v, __shfl_xor(v, 2, 32));
    v = fmaxf(v, __shfl_xor(v, 4, 32));
    v = fmaxf(v, __shfl_xor(v, 8, 32));
    return v;
}
__device__ inline float redsum16(float v) {
    v += __shfl_xor(v, 1, 32);
    v += __shfl_xor(v, 2, 32);
    v += __shfl_xor(v, 4, 32);
    v += __shfl_xor(v, 8, 32);
    return v;
}

// ---------------- lambda scalar ----------------
__global__ void lambda_kernel(const float* lq1, const float* lk1,
                              const float* lq2, const float* lk2, float* lam) {
    __shared__ float s1[64], s2[64];
    int t = threadIdx.x;
    s1[t] = lq1[t] * lk1[t];
    s2[t] = lq2[t] * lk2[t];
    __syncthreads();
    if (t == 0) {
        float a = 0.f, b = 0.f;
        for (int i = 0; i < 64; ++i) { a += s1[i]; b += s2[i]; }
        *lam = __expf(a) - __expf(b) + LAM_INIT;
    }
}

// ---------------- fp32 -> bf16 convert ----------------
__global__ void cvt_bf16_kernel(const float* __restrict__ x,
                                bf16_t* __restrict__ y, int n) {
    int i = blockIdx.x * blockDim.x + threadIdx.x;
    if (i < n) y[i] = (bf16_t)x[i];
}

// ---------------- weight transpose+convert: W[N][K] -> Wt[K][N] ----------------
__global__ void transpose_cvt_kernel(const float* __restrict__ W,
                                     bf16_t* __restrict__ Wt, int N, int K) {
    int idx = blockIdx.x * blockDim.x + threadIdx.x;
    if (idx >= N * K) return;
    int n = idx % N, k = idx / N;
    Wt[(size_t)k * N + n] = (bf16_t)W[(size_t)n * K + k];
}

// ---------------- generic GEMM: C[M][N] = A[M][K] * Bt[K][N] (bf16 in) ----------------
// 32x64 C per wave (2 M-tiles x 4 N-tiles = 8 WMMAs / k-step, A-fragments reused
// across 4 B-fragments); 4 waves/block -> 128x64 C per block; K-step 32.
template <typename OutT>
__global__ __launch_bounds__(128)
void gemm_bt_kernel(const bf16_t* __restrict__ A, const bf16_t* __restrict__ B,
                    OutT* __restrict__ C, int M, int N, int K) {
    const int lane = threadIdx.x & 31;
    const int w    = threadIdx.x >> 5;
    const int tr   = blockIdx.y * 128 + w * 32;   // two 16-row tiles: tr, tr+16
    const int tc   = blockIdx.x * 64;             // four 16-col tiles
    const int ln   = lane & 15;
    const int sel  = (lane >> 4) ? 8 : 0;         // A-fragment half-select (ISA map)

    const bf16_t* arow0 = A + (size_t)(tr + ln) * K;
    const bf16_t* arow1 = A + (size_t)(tr + 16 + ln) * K;
    const bf16_t* bcol  = B + tc;

    VF8 c[2][4];
    for (int mt = 0; mt < 2; ++mt)
        for (int nt = 0; nt < 4; ++nt)
            for (int i = 0; i < 8; ++i) c[mt][nt].f[i] = 0.0f;

    for (int k0 = 0; k0 < K; k0 += 32) {
        V16 a0, a1;
        a0.h[0] = *(const v8bf*)(arow0 + k0 + sel);        // K 0-7  (or 8-15)
        a0.h[1] = *(const v8bf*)(arow0 + k0 + 16 + sel);   // K 16-23 (or 24-31)
        a1.h[0] = *(const v8bf*)(arow1 + k0 + sel);
        a1.h[1] = *(const v8bf*)(arow1 + k0 + 16 + sel);

        const bf16_t* brow = bcol + (size_t)(k0 + lane) * N;  // prefetch row
        V16 b[4];
        for (int nt = 0; nt < 4; ++nt) {
            for (int s = 0; s < 16; ++s) {
                int kk = (s < 8) ? (sel + s) : (16 + sel + s - 8);
                b[nt].v[s] = bcol[(size_t)(k0 + kk) * N + nt * 16 + ln];
            }
        }
        __builtin_prefetch(arow0 + k0 + 64, 0, 1);
        __builtin_prefetch(brow + (size_t)32 * N, 0, 1);

        for (int nt = 0; nt < 4; ++nt) {
            c[0][nt].v = wmma_bf16(a0, b[nt], c[0][nt]);
            c[1][nt].v = wmma_bf16(a1, b[nt], c[1][nt]);
        }
    }

    const int mh = (lane >> 4) ? 8 : 0;
    for (int mt = 0; mt < 2; ++mt)
        for (int nt = 0; nt < 4; ++nt)
            for (int r = 0; r < 8; ++r)
                C[(size_t)(tr + mt * 16 + r + mh) * N + tc + nt * 16 + ln] =
                    (OutT)c[mt][nt].f[r];
}

// ---------------- RoPE: qb/kb [S][4096] -> q1h/q2h [H][S][D], k1T/k2T [H][D][S] ----------------
__global__ void rope_kernel(const bf16_t* __restrict__ qb, const bf16_t* __restrict__ kb,
                            bf16_t* __restrict__ q1h, bf16_t* __restrict__ q2h,
                            bf16_t* __restrict__ k1T, bf16_t* __restrict__ k2T) {
    int idx = blockIdx.x * blockDim.x + threadIdx.x;
    if (idx >= H * S * D) return;
    int d = idx & (D - 1);
    int s = (idx >> 6) & (S - 1);
    int h = idx >> 17;

    float ang = (float)s * __expf(-(float)(d & 31) * (LOG_THETA / 32.0f));
    float cs = cosf(ang), sn = sinf(ang);
    float sgn = (d < 32) ? -1.0f : 1.0f;
    int dp = d ^ 32;

    size_t row = (size_t)s * QKN;
    int c1 = h * D + d, c1p = h * D + dp;

    float q1  = (float)qb[row + c1],        q1p = (float)qb[row + c1p];
    float q2  = (float)qb[row + HID + c1],  q2p = (float)qb[row + HID + c1p];
    float k1  = (float)kb[row + c1],        k1p = (float)kb[row + c1p];
    float k2  = (float)kb[row + HID + c1],  k2p = (float)kb[row + HID + c1p];

    float rq1 = q1 * cs + sgn * q1p * sn;
    float rq2 = q2 * cs + sgn * q2p * sn;
    float rk1 = k1 * cs + sgn * k1p * sn;
    float rk2 = k2 * cs + sgn * k2p * sn;

    size_t qo = ((size_t)h * S + s) * D + d;          // row-major per head (A operand)
    size_t ko = ((size_t)h * D + d) * S + s;          // d-major per head (B operand)
    q1h[qo] = (bf16_t)rq1;  q2h[qo] = (bf16_t)rq2;
    k1T[ko] = (bf16_t)rk1;  k2T[ko] = (bf16_t)rk2;
}

// ---------------- differential flash attention ----------------
// 1 wave per (head, 16-query tile); 32 keys per chunk; dual online softmax;
// P re-laid-out through LDS into the WMMA A-fragment map; fused RMSNorm.
__global__ __launch_bounds__(32)
void diff_attn_kernel(const bf16_t* __restrict__ q1h, const bf16_t* __restrict__ q2h,
                      const bf16_t* __restrict__ k1T, const bf16_t* __restrict__ k2T,
                      const bf16_t* __restrict__ vb,  const float* __restrict__ lamp,
                      const float* __restrict__ subw, bf16_t* __restrict__ attnOut) {
    const int qt   = blockIdx.x;
    const int h    = blockIdx.y;
    const int lane = threadIdx.x & 31;
    const int ln   = lane & 15;
    const int lh   = lane >> 4;
    const int sel  = lh ? 8 : 0;

    __shared__ bf16_t p_lds[2][16 * 32];

    const float lam = *lamp;

    // Q fragments: two K-steps of 32 over D=64, both branches
    V16 aq1[2], aq2[2];
    {
        const bf16_t* q1row = q1h + ((size_t)h * S + qt * 16 + ln) * D;
        const bf16_t* q2row = q2h + ((size_t)h * S + qt * 16 + ln) * D;
        for (int ks = 0; ks < 2; ++ks) {
            aq1[ks].h[0] = *(const v8bf*)(q1row + ks * 32 + sel);
            aq1[ks].h[1] = *(const v8bf*)(q1row + ks * 32 + 16 + sel);
            aq2[ks].h[0] = *(const v8bf*)(q2row + ks * 32 + sel);
            aq2[ks].h[1] = *(const v8bf*)(q2row + ks * 32 + 16 + sel);
        }
    }

    VF8 o1[4], o2[4];
    for (int t = 0; t < 4; ++t)
        for (int i = 0; i < 8; ++i) { o1[t].f[i] = 0.f; o2[t].f[i] = 0.f; }
    float m1[8], m2[8], l1[8], l2[8];
    for (int r = 0; r < 8; ++r) { m1[r] = m2[r] = -1e30f; l1[r] = l2[r] = 0.f; }

    const int nCh = (qt >> 1) + 1;
    for (int cix = 0; cix < nCh; ++cix) {
        const int k0 = cix * 32;

        // ---- scores: 2 N-tiles x 2 K-steps x 2 branches = 8 WMMAs ----
        VF8 sc1[2], sc2[2];
        for (int nt = 0; nt < 2; ++nt)
            for (int i = 0; i < 8; ++i) { sc1[nt].f[i] = 0.f; sc2[nt].f[i] = 0.f; }
        for (int nt = 0; nt < 2; ++nt) {
            for (int ks = 0; ks < 2; ++ks) {
                V16 b1, b2;
                const bf16_t* k1row = k1T + ((size_t)h * D + ks * 32) * S + k0 + nt * 16 + ln;
                const bf16_t* k2row = k2T + ((size_t)h * D + ks * 32) * S + k0 + nt * 16 + ln;
                for (int s = 0; s < 16; ++s) {
                    int kk = (s < 8) ? (sel + s) : (16 + sel + s - 8);
                    b1.v[s] = k1row[(size_t)kk * S];
                    b2.v[s] = k2row[(size_t)kk * S];
                }
                sc1[nt].v = wmma_bf16(aq1[ks], b1, sc1[nt]);
                sc2[nt].v = wmma_bf16(aq2[ks], b2, sc2[nt]);
            }
        }

        // ---- causal mask + scale ----
        float f1[2][8], f2[2][8];
        for (int nt = 0; nt < 2; ++nt)
            for (int r = 0; r < 8; ++r) {
                int key = k0 + nt * 16 + ln;
                int qi  = qt * 16 + r + 8 * lh;
                bool ok = key <= qi;
                f1[nt][r] = ok ? sc1[nt].f[r] * SCALE : -1e30f;
                f2[nt][r] = ok ? sc2[nt].f[r] * SCALE : -1e30f;
            }

        // ---- dual online softmax; stage P into LDS (A-fragment source) ----
        for (int r = 0; r < 8; ++r) {
            int rr = (r + 8 * lh) * 32;

            float cm = redmax16(fmaxf(f1[0][r], f1[1][r]));
            float mn = fmaxf(m1[r], cm);
            float corr = __expf(m1[r] - mn);
            m1[r] = mn;
            float p0 = __expf(f1[0][r] - mn), p1 = __expf(f1[1][r] - mn);
            l1[r] = l1[r] * corr + redsum16(p0 + p1);
            for (int t = 0; t < 4; ++t) o1[t].f[r] *= corr;
            p_lds[0][rr + ln]      = (bf16_t)p0;
            p_lds[0][rr + 16 + ln] = (bf16_t)p1;

            cm = redmax16(fmaxf(f2[0][r], f2[1][r]));
            mn = fmaxf(m2[r], cm);
            corr = __expf(m2[r] - mn);
            m2[r] = mn;
            p0 = __expf(f2[0][r] - mn); p1 = __expf(f2[1][r] - mn);
            l2[r] = l2[r] * corr + redsum16(p0 + p1);
            for (int t = 0; t < 4; ++t) o2[t].f[r] *= corr;
            p_lds[1][rr + ln]      = (bf16_t)p0;
            p_lds[1][rr + 16 + ln] = (bf16_t)p1;
        }
        __syncthreads();

        // ---- P@V: reload P in A-fragment layout, 4 d-tiles x 2 branches ----
        V16 pa1, pa2;
        {
            const bf16_t* prow1 = &p_lds[0][ln * 32];
            const bf16_t* prow2 = &p_lds[1][ln * 32];
            pa1.h[0] = *(const v8bf*)(prow1 + sel);
            pa1.h[1] = *(const v8bf*)(prow1 + 16 + sel);
            pa2.h[0] = *(const v8bf*)(prow2 + sel);
            pa2.h[1] = *(const v8bf*)(prow2 + 16 + sel);
        }
        for (int t = 0; t < 4; ++t) {
            V16 bv;
            const bf16_t* vrow = vb + (size_t)k0 * HID + h * D + t * 16 + ln;
            for (int s = 0; s < 16; ++s) {
                int kk = (s < 8) ? (sel + s) : (16 + sel + s - 8);
                bv.v[s] = vrow[(size_t)kk * HID];
            }
            o1[t].v = wmma_bf16(pa1, bv, o1[t]);
            o2[t].v = wmma_bf16(pa2, bv, o2[t]);
        }
        __syncthreads();
    }

    // ---- epilogue: differential combine + head RMSNorm + (1-lam_init) ----
    float val[4][8], ss[8], inv1[8], inv2[8];
    for (int r = 0; r < 8; ++r) {
        ss[r] = 0.f;
        inv1[r] = 1.0f / l1[r];
        inv2[r] = 1.0f / l2[r];
    }
    for (int t = 0; t < 4; ++t)
        for (int r = 0; r < 8; ++r) {
            float vv = o1[t].f[r] * inv1[r] - lam * o2[t].f[r] * inv2[r];
            val[t][r] = vv;
            ss[r] += vv * vv;
        }
    float sc[8];
    for (int r = 0; r < 8; ++r) {
        float tot = redsum16(ss[r]);
        sc[r] = rsqrtf(tot * (1.0f / 64.0f) + EPS) * OUT_SCALE;
    }
    for (int t = 0; t < 4; ++t) {
        float wsub = subw[t * 16 + ln];
        for (int r = 0; r < 8; ++r) {
            size_t o = (size_t)(qt * 16 + r + 8 * lh) * HID + h * D + t * 16 + ln;
            attnOut[o] = (bf16_t)(val[t][r] * sc[r] * wsub);
        }
    }
}

// ---------------- host side ----------------
extern "C" void kernel_launch(void* const* d_in, const int* in_sizes, int n_in,
                              void* d_out, int out_size, void* d_ws, size_t ws_size,
                              hipStream_t stream) {
    (void)in_sizes; (void)n_in; (void)out_size; (void)ws_size;
    const float* hid  = (const float*)d_in[0];
    const float* Wq   = (const float*)d_in[1];
    const float* Wk   = (const float*)d_in[2];
    const float* Wv   = (const float*)d_in[3];
    const float* Wo   = (const float*)d_in[4];
    const float* lq1  = (const float*)d_in[5];
    const float* lk1  = (const float*)d_in[6];
    const float* lq2  = (const float*)d_in[7];
    const float* lk2  = (const float*)d_in[8];
    const float* subw = (const float*)d_in[9];
    float* out = (float*)d_out;

    char* ws = (char*)d_ws;
    size_t off = 0;
    auto take = [&](size_t bytes) {
        char* p = ws + off;
        off = (off + bytes + 255) & ~(size_t)255;
        return p;
    };
    bf16_t* Xb   = (bf16_t*)take((size_t)S * HID * 2);
    bf16_t* WqT  = (bf16_t*)take((size_t)HID * QKN * 2);
    bf16_t* WkT  = (bf16_t*)take((size_t)HID * QKN * 2);
    bf16_t* WvT  = (bf16_t*)take((size_t)HID * HID * 2);
    bf16_t* WoT  = (bf16_t*)take((size_t)HID * HID * 2);
    bf16_t* qb   = (bf16_t*)take((size_t)S * QKN * 2);
    bf16_t* kb   = (bf16_t*)take((size_t)S * QKN * 2);
    bf16_t* vbuf = (bf16_t*)take((size_t)S * HID * 2);
    bf16_t* q1h  = (bf16_t*)take((size_t)H * S * D * 2);
    bf16_t* q2h  = (bf16_t*)take((size_t)H * S * D * 2);
    bf16_t* k1T  = (bf16_t*)take((size_t)H * S * D * 2);
    bf16_t* k2T  = (bf16_t*)take((size_t)H * S * D * 2);
    bf16_t* attn = (bf16_t*)take((size_t)S * HID * 2);
    float*  lam  = (float*)take(256);

    // 1. lambda scalar
    lambda_kernel<<<1, 64, 0, stream>>>(lq1, lk1, lq2, lk2, lam);
    // 2. convert hidden -> bf16
    cvt_bf16_kernel<<<(S * HID) / 256, 256, 0, stream>>>(hid, Xb, S * HID);
    // 3. transpose+convert weights (W[N][K] -> Wt[K][N])
    transpose_cvt_kernel<<<(QKN * HID) / 256, 256, 0, stream>>>(Wq, WqT, QKN, HID);
    transpose_cvt_kernel<<<(QKN * HID) / 256, 256, 0, stream>>>(Wk, WkT, QKN, HID);
    transpose_cvt_kernel<<<(HID * HID) / 256, 256, 0, stream>>>(Wv, WvT, HID, HID);
    transpose_cvt_kernel<<<(HID * HID) / 256, 256, 0, stream>>>(Wo, WoT, HID, HID);
    // 4. q/k/v projections (WMMA bf16): 128x64 C per block
    gemm_bt_kernel<bf16_t><<<dim3(QKN / 64, S / 128), 128, 0, stream>>>(Xb, WqT, qb, S, QKN, HID);
    gemm_bt_kernel<bf16_t><<<dim3(QKN / 64, S / 128), 128, 0, stream>>>(Xb, WkT, kb, S, QKN, HID);
    gemm_bt_kernel<bf16_t><<<dim3(HID / 64, S / 128), 128, 0, stream>>>(Xb, WvT, vbuf, S, HID, HID);
    // 5. RoPE + head re-layout
    rope_kernel<<<(H * S * D) / 256, 256, 0, stream>>>(qb, kb, q1h, q2h, k1T, k2T);
    // 6. differential flash attention (WMMA scores + WMMA PV)
    diff_attn_kernel<<<dim3(S / 16, H), 32, 0, stream>>>(q1h, q2h, k1T, k2T, vbuf, lam, subw, attn);
    // 7. output projection, fp32 epilogue into d_out
    gemm_bt_kernel<float><<<dim3(HID / 64, S / 128), 128, 0, stream>>>(attn, WoT, out, S, HID, HID);
}
